// MultiplicativeMLPLayer_87668872446208
// MI455X (gfx1250) — matrix operand order repaired and checked by reading, not compiled, not run on hardware
//
#include <hip/hip_runtime.h>

// ---------------------------------------------------------------------------
// Problem constants (from reference): B=1, S=256, D=512, P=4096, TOPK=8
// ---------------------------------------------------------------------------
#define DD    512
#define SS    256
#define PP    4096
#define KSEL  8
#define HMAX  2560
#define NEXP  16
#define NROWS (SS*KSEL)   // 2048 (token,slot) rows

typedef __attribute__((ext_vector_type(16))) __bf16 v16bf;
typedef __attribute__((ext_vector_type(8)))  __bf16 v8bf;
typedef __attribute__((ext_vector_type(8)))  float  v8f;
typedef __attribute__((ext_vector_type(2)))  int    v2i;

__device__ __forceinline__ float gelu_erf(float x) {
    return 0.5f * x * (1.0f + erff(x * 0.70710678118654752f));
}
__device__ __forceinline__ int imin(int a, int b) { return a < b ? a : b; }

// ------------------- async global->LDS staging (gfx1250) -------------------
#if defined(__has_builtin)
# if __has_builtin(__builtin_amdgcn_global_load_async_to_lds_b64)
#  define HAVE_ASYNC_LDS 1
# else
#  define HAVE_ASYNC_LDS 0
# endif
#else
# define HAVE_ASYNC_LDS 0
#endif

__device__ __forceinline__ void copy8_to_lds(__bf16* dst, const __bf16* src) {
#if HAVE_ASYNC_LDS
    __builtin_amdgcn_global_load_async_to_lds_b64(
        (v2i __attribute__((address_space(1)))*)(src),
        (v2i __attribute__((address_space(3)))*)(dst), 0, 0);
#else
    *(unsigned long long*)dst = *(const unsigned long long*)src;
#endif
}
__device__ __forceinline__ void wait_lds_copies() {
#if HAVE_ASYNC_LDS
# if __has_builtin(__builtin_amdgcn_s_wait_asynccnt)
    __builtin_amdgcn_s_wait_asynccnt(0);
# else
    asm volatile("s_wait_asynccnt 0" ::: "memory");
# endif
#endif
}

// ------------------------- WMMA fragment accessors -------------------------
// A fragment (16x32 bf16) read from a [16 rows x 32 K] LDS tile:
// lane row M = lane&15; K base kb=(lane>>4)*16; elems 0..15 -> kb..kb+15
__device__ __forceinline__ v16bf read_a_lds(const __bf16* atile, int lane) {
    const __bf16* p = atile + (lane & 15) * 32 + ((lane >> 4) << 4);
    v8bf lo = *(const v8bf*)p;
    v8bf hi = *(const v8bf*)(p + 8);
    return __builtin_shufflevector(lo, hi, 0, 1, 2, 3, 4, 5, 6, 7,
                                           8, 9, 10, 11, 12, 13, 14, 15);
}
// B fragment (32x16 bf16) from transposed bf16 weights Bt[N][K]:
// lane col n; K base = (lane>>4)*16; 16 consecutive K values = 32 contiguous B.
__device__ __forceinline__ v16bf read_b_bt(const __bf16* btn /* Bt + n*K */, int k0, int lane) {
    return *(const v16bf*)(btn + k0 + ((lane >> 4) << 4));
}

// ---------------------------------------------------------------------------
// Weight transpose+cast: out[z][n][k] = bf16(in[z][k][n]).  Block 32x8.
// ---------------------------------------------------------------------------
__global__ __launch_bounds__(256) void transpose_cast_kernel(
    const float* in, __bf16* out, int K, int N, long sInZ, long sOutZ)
{
    __shared__ float tile[32][33];
    int z = blockIdx.z;
    const float* inz = in + (size_t)z * sInZ;
    __bf16* outz = out + (size_t)z * sOutZ;
    int n0 = blockIdx.x * 32, k0 = blockIdx.y * 32;
    int tx = threadIdx.x, ty = threadIdx.y;
#pragma unroll
    for (int r = 0; r < 32; r += 8)
        tile[ty + r][tx] = inz[(size_t)(k0 + ty + r) * N + n0 + tx];
    __syncthreads();
#pragma unroll
    for (int r = 0; r < 32; r += 8)
        outz[(size_t)(n0 + ty + r) * K + k0 + tx] = (__bf16)tile[tx][ty + r];
}

__global__ __launch_bounds__(256) void cast_bf16_kernel(const float* in, __bf16* out, int n) {
    int i = blockIdx.x * 256 + threadIdx.x;
    if (i < n) out[i] = (__bf16)in[i];
}

// ---------------------------------------------------------------------------
// Batched GEMM, contiguous rows. Block = 128 threads (4 waves).
// All 4 waves share one 16x32 A-tile via LDS (async-staged); each wave owns a
// 16-wide N strip, so the block computes a 16x64 C macro-tile per (bx,by,z).
// ---------------------------------------------------------------------------
__global__ __launch_bounds__(128) void gemm_bf16t_kernel(
    const __bf16* A, long sAz, int lda,
    const __bf16* Bt, long sBz,
    const float* bias, long sbz,
    float* outF, __bf16* outB, long sOz, int ldo,
    int K, int act)
{
    __shared__ __bf16 atile[16 * 32];
    int tid = threadIdx.x;
    int lane = tid & 31, w = tid >> 5;
    int z = blockIdx.z;
    int m0 = blockIdx.y * 16;
    int n = blockIdx.x * 64 + w * 16 + (lane & 15);
    const __bf16* Az  = A + (size_t)z * sAz;
    const __bf16* btn = Bt + (size_t)z * sBz + (size_t)n * K;
    int arow = tid >> 3, aseg = tid & 7;                   // 16 rows x 8 segs of 4
    const __bf16* agsrc = Az + (size_t)(m0 + arow) * lda + aseg * 4;
    __bf16* adst = atile + arow * 32 + aseg * 4;
    v8f c = {};
    for (int k0 = 0; k0 < K; k0 += 32) {
        __syncthreads();
        copy8_to_lds(adst, agsrc + k0);
        if (k0 + 32 < K) __builtin_prefetch(btn + k0 + 32, 0, 1);
        wait_lds_copies();
        __syncthreads();
        v16bf a = read_a_lds(atile, lane);
        v16bf b = read_b_bt(btn, k0, lane);
        c = __builtin_amdgcn_wmma_f32_16x16x32_bf16(false, a, false, b, (short)0, c, false, false);
    }
    float bb = bias ? bias[(size_t)z * sbz + n] : 0.0f;
#pragma unroll
    for (int r = 0; r < 8; ++r) {
        int row = m0 + r + ((lane >> 4) << 3);
        float v = c[r] + bb;
        if (act == 1) v = gelu_erf(v);
        size_t o = (size_t)z * sOz + (size_t)row * ldo + n;
        if (outF) outF[o] = v;
        if (outB) outB[o] = (__bf16)v;
    }
}

// ---------------------------------------------------------------------------
// Batched GEMM over expert-compacted (gathered) rows. Same 4-wave structure.
// z = expert. act: 0 none, 3 = gelu/relu by expert parity.
// useHdForK / useHdForN: use hd = D*(2+z/4) as K length / N limit.
// ---------------------------------------------------------------------------
__global__ __launch_bounds__(128) void gemm_gather_kernel(
    const __bf16* A, int lda,
    const __bf16* Bt, long sBz,
    const float* bias, long sbz,
    const int* rows, const int* cnt,
    float* outF, __bf16* outB, int ldo,
    int K, int act, int useHdForK, int useHdForN)
{
    __shared__ __bf16 atile[16 * 32];
    __shared__ int srow[16];
    int z = blockIdx.z;
    int hd = DD * (2 + (z >> 2));
    if (useHdForN && blockIdx.x * 64 >= hd) return;        // uniform per block
    int ce = cnt[z];
    int m0 = blockIdx.y * 16;
    if (m0 >= ce) return;                                  // uniform per block
    if (useHdForK) K = hd;

    int tid = threadIdx.x;
    int lane = tid & 31, w = tid >> 5;
    int n = blockIdx.x * 64 + w * 16 + (lane & 15);
    if (tid < 16) srow[tid] = rows[z * NROWS + imin(m0 + tid, ce - 1)];
    __syncthreads();

    const __bf16* btn = Bt + (size_t)z * sBz + (size_t)n * K;
    int arow = tid >> 3, aseg = tid & 7;
    const __bf16* agsrc = A + (size_t)srow[arow] * lda + aseg * 4;
    __bf16* adst = atile + arow * 32 + aseg * 4;
    v8f c = {};
    for (int k0 = 0; k0 < K; k0 += 32) {
        __syncthreads();
        copy8_to_lds(adst, agsrc + k0);
        if (k0 + 32 < K) __builtin_prefetch(btn + k0 + 32, 0, 1);
        wait_lds_copies();
        __syncthreads();
        v16bf a = read_a_lds(atile, lane);
        v16bf b = read_b_bt(btn, k0, lane);
        c = __builtin_amdgcn_wmma_f32_16x16x32_bf16(false, a, false, b, (short)0, c, false, false);
    }
    float bb = bias ? bias[(size_t)z * sbz + n] : 0.0f;
    int geluSel = (act == 3) ? ((z & 1) == 0) : 0;
#pragma unroll
    for (int r = 0; r < 8; ++r) {
        int local = r + ((lane >> 4) << 3);
        if (m0 + local < ce) {
            int rid = srow[local];
            float v = c[r] + bb;
            if (act == 3) v = geluSel ? gelu_erf(v) : fmaxf(v, 0.0f);
            size_t o = (size_t)rid * ldo + n;
            if (outF) outF[o] = v;
            if (outB) outB[o] = (__bf16)v;
        }
    }
}

// ---------------------------------------------------------------------------
// Per-token softmax, freq accumulation, top-8, pathway decomposition,
// per-expert row compaction, and pw output. One 256-thread block per token.
// ---------------------------------------------------------------------------
__global__ __launch_bounds__(256) void softmax_topk_kernel(
    const float* scores, const float* temp, float* freq,
    float* top_vals, int* top_idx, int* pre_i, int* mlp_i, int* post_i,
    int* mlp_cnt, int* mlp_rows, int* post_cnt, int* post_rows, float* pw)
{
    __shared__ float sm[PP];
    __shared__ float red[256];
    __shared__ int   redi[256];
    __shared__ float selv[KSEL];
    __shared__ int   seli[KSEL];
    __shared__ float wsum_s;
    int t = blockIdx.x, tid = threadIdx.x;
    const float* srowp = scores + (size_t)t * PP;
    float tinv = 1.0f / temp[0];

    float mx = -3.4e38f;
    for (int p = tid; p < PP; p += 256) mx = fmaxf(mx, srowp[p]);
    red[tid] = mx; __syncthreads();
    for (int s = 128; s; s >>= 1) { if (tid < s) red[tid] = fmaxf(red[tid], red[tid + s]); __syncthreads(); }
    float m = red[0]; __syncthreads();

    float s1 = 0.f, s2 = 0.f;
    for (int p = tid; p < PP; p += 256) {
        float d  = srowp[p] - m;
        float e1 = expf(d);
        float e2 = expf(d * tinv);
        sm[p] = e2;
        s1 += e1; s2 += e2;
    }
    red[tid] = s1; __syncthreads();
    for (int s = 128; s; s >>= 1) { if (tid < s) red[tid] += red[tid + s]; __syncthreads(); }
    s1 = red[0]; __syncthreads();
    red[tid] = s2; __syncthreads();
    for (int s = 128; s; s >>= 1) { if (tid < s) red[tid] += red[tid + s]; __syncthreads(); }
    s2 = red[0]; __syncthreads();
    float rs1 = 1.0f / s1, rs2 = 1.0f / s2;

    for (int p = tid; p < PP; p += 256) {
        atomicAdd(&freq[p], expf(srowp[p] - m) * rs1 * (1.0f / (float)SS));
        sm[p] *= rs2;
    }
    __syncthreads();

    for (int k = 0; k < KSEL; ++k) {
        float bv = -1.0f; int bi = PP;
        for (int p = tid; p < PP; p += 256) {
            float v = sm[p];
            if (v > bv || (v == bv && p < bi)) { bv = v; bi = p; }
        }
        red[tid] = bv; redi[tid] = bi; __syncthreads();
        for (int s = 128; s; s >>= 1) {
            if (tid < s) {
                if (red[tid + s] > red[tid] ||
                    (red[tid + s] == red[tid] && redi[tid + s] < redi[tid])) {
                    red[tid] = red[tid + s]; redi[tid] = redi[tid + s];
                }
            }
            __syncthreads();
        }
        if (tid == 0) { selv[k] = red[0]; seli[k] = redi[0]; sm[redi[0]] = -1.0f; }
        __syncthreads();
    }

    if (tid == 0) {
        float wv = 0.f;
        for (int k = 0; k < KSEL; ++k) wv += selv[k];
        wsum_s = wv;
    }
    float* pwrow = pw + (size_t)t * PP;
    for (int p = tid; p < PP; p += 256) pwrow[p] = 0.0f;
    __syncthreads();

    if (tid < KSEL) {
        int k = tid;
        int r = t * KSEL + k;
        float v = selv[k]; int idx = seli[k];
        top_vals[r] = v; top_idx[r] = idx;
        int pre = idx / 256, rem = idx % 256;
        int mlp = rem / 16, post = rem % 16;
        pre_i[r] = pre; mlp_i[r] = mlp; post_i[r] = post;
        int pm = atomicAdd(&mlp_cnt[mlp], 1);
        mlp_rows[mlp * NROWS + pm] = r;
        int pq = atomicAdd(&post_cnt[post], 1);
        post_rows[post * NROWS + pq] = r;
        pwrow[idx] = v / (wsum_s + 1e-8f);
    }
}

__global__ __launch_bounds__(256) void glbl_kernel(const float* freq, float* loss) {
    __shared__ float s1[256], s2[256];
    int tid = threadIdx.x;
    float a = 0.f, b = 0.f;
    for (int p = tid; p < PP; p += 256) { float f = freq[p]; a += f; b += f * f; }
    s1[tid] = a; s2[tid] = b; __syncthreads();
    for (int s = 128; s; s >>= 1) { if (tid < s) { s1[tid] += s1[tid + s]; s2[tid] += s2[tid + s]; } __syncthreads(); }
    if (tid == 0) {
        float mean = s1[0] / (float)PP;
        float var  = (s2[0] - (float)PP * mean * mean) / (float)(PP - 1);
        *loss = (float)PP * var;
    }
}

// Gather selected pre-expert row, LayerNorm, activation (e%3), write bf16 x_pre.
__global__ __launch_bounds__(256) void preact_kernel(
    const float* pre_lin, const int* pre_i, const float* gpre, const float* bpre_ln,
    __bf16* x_pre)
{
    __shared__ float s1[256], s2[256];
    int r = blockIdx.x, tid = threadIdx.x;
    int t = r >> 3;
    int e = pre_i[r];
    const float* row = pre_lin + ((size_t)e * SS + t) * DD;
    float v0 = row[tid], v1 = row[tid + 256];
    s1[tid] = v0 + v1; s2[tid] = v0 * v0 + v1 * v1; __syncthreads();
    for (int s = 128; s; s >>= 1) { if (tid < s) { s1[tid] += s1[tid + s]; s2[tid] += s2[tid + s]; } __syncthreads(); }
    float mean = s1[0] * (1.0f / DD);
    float var  = s2[0] * (1.0f / DD) - mean * mean;
    float rstd = rsqrtf(var + 1e-5f);
    int aid = e % 3;
#pragma unroll
    for (int h = 0; h < 2; ++h) {
        int col = tid + h * 256;
        float v = (h ? v1 : v0);
        v = (v - mean) * rstd * gpre[(size_t)e * DD + col] + bpre_ln[(size_t)e * DD + col];
        if (aid == 0)      v = gelu_erf(v);
        else if (aid == 1) v = fmaxf(v, 0.0f);
        else               v = tanhf(v);
        x_pre[(size_t)r * DD + col] = (__bf16)v;
    }
}

// Finalize: optional post LN (p%2==0), weighted sum over 8 slots -> output.
__global__ __launch_bounds__(256) void finalize_kernel(
    const float* post_lin, const int* post_i, const float* top_vals,
    const float* gpost, const float* bpost_ln, float* out)
{
    __shared__ float s1[256], s2[256];
    int t = blockIdx.x, tid = threadIdx.x;
    float acc0 = 0.f, acc1 = 0.f;
    for (int k = 0; k < KSEL; ++k) {
        int r = t * KSEL + k;
        int p = post_i[r];
        float w = top_vals[r];
        const float* row = post_lin + (size_t)r * DD;
        float v0 = row[tid], v1 = row[tid + 256];
        if ((p & 1) == 0) {
            s1[tid] = v0 + v1; s2[tid] = v0 * v0 + v1 * v1; __syncthreads();
            for (int s = 128; s; s >>= 1) { if (tid < s) { s1[tid] += s1[tid + s]; s2[tid] += s2[tid + s]; } __syncthreads(); }
            float mean = s1[0] * (1.0f / DD);
            float var  = s2[0] * (1.0f / DD) - mean * mean;
            float rstd = rsqrtf(var + 1e-5f);
            __syncthreads();
            v0 = (v0 - mean) * rstd * gpost[(size_t)p * DD + tid]       + bpost_ln[(size_t)p * DD + tid];
            v1 = (v1 - mean) * rstd * gpost[(size_t)p * DD + tid + 256] + bpost_ln[(size_t)p * DD + tid + 256];
        }
        acc0 += w * v0; acc1 += w * v1;
    }
    out[(size_t)t * DD + tid]       = acc0;
    out[(size_t)t * DD + tid + 256] = acc1;
}

// ---------------------------------------------------------------------------
extern "C" void kernel_launch(void* const* d_in, const int* in_sizes, int n_in,
                              void* d_out, int out_size, void* d_ws, size_t ws_size,
                              hipStream_t stream)
{
    const float* x        = (const float*)d_in[0];
    const float* Wr1      = (const float*)d_in[1];
    const float* br1      = (const float*)d_in[2];
    const float* Wr2      = (const float*)d_in[3];
    const float* br2      = (const float*)d_in[4];
    const float* Wr3      = (const float*)d_in[5];
    const float* br3      = (const float*)d_in[6];
    const float* temp     = (const float*)d_in[7];
    const float* Wpre     = (const float*)d_in[8];
    const float* bpre     = (const float*)d_in[9];
    const float* gpre     = (const float*)d_in[10];
    const float* bpre_ln  = (const float*)d_in[11];
    const float* W1       = (const float*)d_in[12];
    const float* b1       = (const float*)d_in[13];
    const float* W2       = (const float*)d_in[14];
    const float* b2       = (const float*)d_in[15];
    const float* Wpost    = (const float*)d_in[16];
    const float* bpost    = (const float*)d_in[17];
    const float* gpost    = (const float*)d_in[18];
    const float* bpost_ln = (const float*)d_in[19];

    float* out  = (float*)d_out;          // [256,512]
    float* loss = out + SS * DD;          // scalar
    float* pw   = loss + 1;               // [256,4096]

    char* ws = (char*)d_ws;
    size_t off = 0;
    auto take = [&](size_t bytes) -> char* {
        char* p = ws + off;
        off = (off + bytes + 255) & ~(size_t)255;
        return p;
    };
    float*  scores   = (float*)take(sizeof(float) * SS * PP);
    float*  freq     = (float*)take(sizeof(float) * PP);
    float*  top_vals = (float*)take(sizeof(float) * NROWS);
    int*    top_idx  = (int*)  take(sizeof(int) * NROWS);
    int*    pre_i    = (int*)  take(sizeof(int) * NROWS);
    int*    mlp_i    = (int*)  take(sizeof(int) * NROWS);
    int*    post_i   = (int*)  take(sizeof(int) * NROWS);
    int*    mlp_cnt  = (int*)  take(sizeof(int) * NEXP);
    int*    post_cnt = (int*)  take(sizeof(int) * NEXP);
    int*    mlp_rows = (int*)  take(sizeof(int) * NEXP * NROWS);
    int*    post_rows= (int*)  take(sizeof(int) * NEXP * NROWS);
    float*  pre_lin  = (float*)take(sizeof(float) * NEXP * SS * DD);
    __bf16* x_pre    = (__bf16*)take(2ull * NROWS * DD);
    __bf16* hbuf     = (__bf16*)take(2ull * NROWS * HMAX);
    __bf16* mlp_o    = (__bf16*)take(2ull * NROWS * DD);
    float*  post_lin = (float*)take(sizeof(float) * NROWS * DD);
    __bf16* h1       = (__bf16*)take(2ull * SS * 256);
    __bf16* h2       = (__bf16*)take(2ull * SS * 128);
    // bf16 transposed weights [N][K] and bf16 x
    __bf16* xb       = (__bf16*)take(2ull * SS * DD);
    __bf16* Wr1t     = (__bf16*)take(2ull * 256 * DD);
    __bf16* Wr2t     = (__bf16*)take(2ull * 128 * 256);
    __bf16* Wr3t     = (__bf16*)take(2ull * PP * 128);
    __bf16* Wpret    = (__bf16*)take(2ull * NEXP * DD * DD);
    __bf16* W1t      = (__bf16*)take(2ull * NEXP * (size_t)HMAX * DD);
    __bf16* W2t      = (__bf16*)take(2ull * NEXP * (size_t)DD * HMAX);
    __bf16* Wpostt   = (__bf16*)take(2ull * NEXP * DD * DD);
    (void)ws_size; (void)in_sizes; (void)n_in; (void)out_size;

    (void)hipMemsetAsync(freq, 0, sizeof(float) * PP, stream);
    (void)hipMemsetAsync(mlp_cnt, 0, sizeof(int) * NEXP, stream);
    (void)hipMemsetAsync(post_cnt, 0, sizeof(int) * NEXP, stream);

    // ---- one-time (per call) bf16 casts / weight transposes ----
    cast_bf16_kernel<<<(SS * DD + 255) / 256, 256, 0, stream>>>(x, xb, SS * DD);
    dim3 tb(32, 8);
    transpose_cast_kernel<<<dim3(256 / 32, DD / 32, 1), tb, 0, stream>>>(Wr1, Wr1t, DD, 256, 0, 0);
    transpose_cast_kernel<<<dim3(128 / 32, 256 / 32, 1), tb, 0, stream>>>(Wr2, Wr2t, 256, 128, 0, 0);
    transpose_cast_kernel<<<dim3(PP / 32, 128 / 32, 1), tb, 0, stream>>>(Wr3, Wr3t, 128, PP, 0, 0);
    transpose_cast_kernel<<<dim3(DD / 32, DD / 32, NEXP), tb, 0, stream>>>(
        Wpre, Wpret, DD, DD, (long)DD * DD, (long)DD * DD);
    transpose_cast_kernel<<<dim3(HMAX / 32, DD / 32, NEXP), tb, 0, stream>>>(
        W1, W1t, DD, HMAX, (long)DD * HMAX, (long)DD * HMAX);
    transpose_cast_kernel<<<dim3(DD / 32, HMAX / 32, NEXP), tb, 0, stream>>>(
        W2, W2t, HMAX, DD, (long)DD * HMAX, (long)DD * HMAX);
    transpose_cast_kernel<<<dim3(DD / 32, DD / 32, NEXP), tb, 0, stream>>>(
        Wpost, Wpostt, DD, DD, (long)DD * DD, (long)DD * DD);

    // ---- router: x@Wr1 (gelu) -> @Wr2 (gelu) -> @Wr3 -> scores ----
    gemm_bf16t_kernel<<<dim3(256 / 64, SS / 16, 1), 128, 0, stream>>>(
        xb, 0, DD, Wr1t, 0, br1, 0, nullptr, h1, 0, 256, DD, 1);
    gemm_bf16t_kernel<<<dim3(128 / 64, SS / 16, 1), 128, 0, stream>>>(
        h1, 0, 256, Wr2t, 0, br2, 0, nullptr, h2, 0, 128, 256, 1);
    gemm_bf16t_kernel<<<dim3(PP / 64, SS / 16, 1), 128, 0, stream>>>(
        h2, 0, 128, Wr3t, 0, br3, 0, scores, nullptr, 0, PP, 128, 0);

    // ---- softmax + top-k + routing tables + pw, then GLBL loss ----
    softmax_topk_kernel<<<SS, 256, 0, stream>>>(
        scores, temp, freq, top_vals, top_idx, pre_i, mlp_i, post_i,
        mlp_cnt, mlp_rows, post_cnt, post_rows, pw);
    glbl_kernel<<<1, 256, 0, stream>>>(freq, loss);

    // ---- pre experts: x @ Wpre[e] + bpre[e] for all 16 experts ----
    gemm_bf16t_kernel<<<dim3(DD / 64, SS / 16, NEXP), 128, 0, stream>>>(
        xb, 0, DD, Wpret, (long)DD * DD, bpre, DD,
        pre_lin, nullptr, (long)SS * DD, DD, DD, 0);
    preact_kernel<<<NROWS, 256, 0, stream>>>(pre_lin, pre_i, gpre, bpre_ln, x_pre);

    // ---- compacted MLP experts ----
    gemm_gather_kernel<<<dim3(HMAX / 64, NROWS / 16, NEXP), 128, 0, stream>>>(
        x_pre, DD, W1t, (long)DD * HMAX, b1, HMAX, mlp_rows, mlp_cnt,
        nullptr, hbuf, HMAX, DD, /*act=*/3, /*KfromHd=*/0, /*NfromHd=*/1);
    gemm_gather_kernel<<<dim3(DD / 64, NROWS / 16, NEXP), 128, 0, stream>>>(
        hbuf, HMAX, W2t, (long)DD * HMAX, b2, DD, mlp_rows, mlp_cnt,
        nullptr, mlp_o, DD, DD, /*act=*/0, /*KfromHd=*/1, /*NfromHd=*/0);

    // ---- compacted post experts -> f32 post_lin ----
    gemm_gather_kernel<<<dim3(DD / 64, NROWS / 16, NEXP), 128, 0, stream>>>(
        mlp_o, DD, Wpostt, (long)DD * DD, bpost, DD, post_rows, post_cnt,
        post_lin, nullptr, DD, DD, /*act=*/0, 0, 0);

    // ---- optional post LN + weighted slot combine -> output ----
    finalize_kernel<<<SS, 256, 0, stream>>>(
        post_lin, post_i, top_vals, gpost, bpost_ln, out);
}
